// RNNBlock_40707700031587
// MI455X (gfx1250) — matrix operand order repaired and checked by reading, not compiled, not run on hardware
//
#include <hip/hip_runtime.h>
#include <hip/hip_bf16.h>

// ---------------------------------------------------------------------------
// RNN with per-timestep weights on gfx1250 (MI455X), bf16 WMMA + f32 accum.
//   Phase 0: transpose/convert Wx, Wh -> bf16 [t][n][k] in d_ws (B-operand layout)
//   Phase 1: A[b][t][:] = x[b][t][:] @ Wx[t]   (parallel batched GEMM -> d_out)
//   Phase 2: h_t = tanh(A[t] + h_{t-1} @ Wh[t]) per 16-row batch tile, with
//            Wh[t] double-buffered in LDS by the Tensor Data Mover (TDM).
// ---------------------------------------------------------------------------

#define S_LEN 512
#define D     256      // D_IN == D_OUT == 256
#define LDA   264      // padded LDS row stride (bf16 elements), 528B pitch
#define LDW   264      // Wh LDS row stride (must match TDM pad: 256 + 8)

typedef __bf16 bf16_t;
typedef __attribute__((ext_vector_type(16))) __bf16 v16bf;
typedef __attribute__((ext_vector_type(8)))  float  v8f;

typedef unsigned int u32x4 __attribute__((ext_vector_type(4)));
typedef int          i32x4 __attribute__((ext_vector_type(4)));
typedef int          i32x8 __attribute__((ext_vector_type(8)));

union V16B { v16bf v; uint4 q[2]; };

#ifndef __has_builtin
#define __has_builtin(x) 0
#endif
#if __has_builtin(__builtin_amdgcn_tensor_load_to_lds) && \
    __has_builtin(__builtin_amdgcn_s_wait_tensorcnt)
#define USE_TDM 1
#else
#define USE_TDM 0
#endif

// ---- WMMA wrapper: D = A(16x32 bf16) * B(32x16 bf16) + C(16x16 f32) --------
static __device__ __forceinline__ v8f wmma_bf16(v16bf a, v16bf b, v8f c) {
    return __builtin_amdgcn_wmma_f32_16x16x32_bf16(
        /*neg_a=*/false, a, /*neg_b=*/false, b,
        /*c_mod=*/(short)0, c, /*reuse_a=*/false, /*reuse_b=*/false);
}

// ---- B operand: WT[n][k] bf16, row stride `ld` --------------------------
// B 32x16 layout used here (contiguous k): lane 0-15 -> col n=lane,  K = kc+0..15
//                            lane 16-31 -> col n=lane-16, K = kc+16..31
static __device__ __forceinline__ v16bf load_b(const bf16_t* __restrict__ WT,
                                               int ld, int colBase, int kc, int lane) {
    int col = colBase + (lane & 15);
    int k0  = kc + ((lane >> 4) << 4);
    const uint4* p = (const uint4*)(WT + (size_t)col * ld + k0);
    V16B b;
    b.q[0] = p[0];
    b.q[1] = p[1];
    return b.v;
}

// ---- A operand from LDS: lds[row][k] bf16, stride LDA ----------------------
// A 16x32 layout (matches load_b): lanes 0-15 row=lane,  K = kc+0..15
//                                  lanes 16-31 row=lane-16, K = kc+16..31
static __device__ __forceinline__ v16bf load_a(const bf16_t* __restrict__ lds,
                                               int rowBase, int kc, int lane) {
    int row = rowBase + (lane & 15);
    int kA  = kc + ((lane >> 4) << 4);
    const bf16_t* p = lds + row * LDA;
    V16B a;
    a.q[0] = *(const uint4*)(p + kA);
    a.q[1] = *(const uint4*)(p + kA + 8);
    return a.v;
}

// ---------------------------------------------------------------------------
// TDM: DMA one t-slice of WhT (256x256 bf16, contiguous 128KB) into LDS with
// 16B padding after each 512B row => LDS row pitch 264 bf16. D# per ISA ch.8.
// Viewed as 2D: tile 8192 elem x 8 rows, stride 8192 (contiguous).
// ---------------------------------------------------------------------------
#if USE_TDM
static __device__ __forceinline__ void tdm_load_wh(const bf16_t* __restrict__ src,
                                                   bf16_t* __restrict__ lds_dst) {
    size_t ga = (size_t)src;
    u32x4 g0;
    g0.x = 1u;                                        // count=1, user descriptor
    g0.y = (unsigned)(size_t)(void*)lds_dst;          // lds_addr (LDS byte offset)
    g0.z = (unsigned)(ga & 0xFFFFFFFFull);            // global_addr[31:0]
    g0.w = (unsigned)((ga >> 32) & 0x1FFFFFFull)      // global_addr[56:32]
         | (2u << 30);                                // type = 2 ("image")
    i32x8 g1;
    g1[0] = (1 << 16)                                 // data_size = 1 (2 bytes)
          | (1 << 20)                                 // pad_enable
          | (6 << 22)                                 // pad_interval: 128 DWORDs (512B)
          | (3 << 25);                                // pad_amount: 4 DWORDs (16B)
    g1[1] = (int)(8192u << 16);                       // tensor_dim0[15:0] = 8192
    g1[2] = (int)(8u << 16);                          // tensor_dim0 hi=0; tensor_dim1=8
    g1[3] = (int)(8192u << 16);                       // tensor_dim1 hi=0; tile_dim0=8192
    g1[4] = 8;                                        // tile_dim1=8, tile_dim2=0
    g1[5] = 8192;                                     // tensor_dim0_stride[31:0]
    g1[6] = 0;                                        // stride hi, dim1_stride lo
    g1[7] = 0;
    i32x4 z4 = {0, 0, 0, 0};
#if defined(__clang_major__) && (__clang_major__ >= 23)
    i32x8 z8 = {0, 0, 0, 0, 0, 0, 0, 0};
    __builtin_amdgcn_tensor_load_to_lds(g0, g1, z4, z4, z8, 0);
#else
    __builtin_amdgcn_tensor_load_to_lds(g0, g1, z4, z4, 0);
#endif
}
#endif

static __device__ __forceinline__ void stage_wh(bf16_t* __restrict__ dst,
                                                const bf16_t* __restrict__ src,
                                                int tid) {
#if USE_TDM
    if ((tid >> 5) == 0) tdm_load_wh(src, dst);       // one TDM issue (wave 0)
#else
    // Cooperative fallback: 8192 16B chunks, row-padded store
    const uint4* s = (const uint4*)src;
    #pragma unroll
    for (int c = tid; c < 8192; c += 256) {
        int row = c >> 4;
        ((uint4*)(dst + row * LDW))[c & 15] = s[c];
    }
#endif
}

static __device__ __forceinline__ void stage_wait(int tid) {
#if USE_TDM
    if ((tid >> 5) == 0) __builtin_amdgcn_s_wait_tensorcnt(0);
#endif
}

// ---------------------------------------------------------------------------
// Phase 0: W [t][k][n] f32  ->  WT [t][n][k] bf16   (32x32 LDS tile transpose)
// ---------------------------------------------------------------------------
__global__ __launch_bounds__(256) void k_transpose_cvt(const float* __restrict__ W,
                                                       bf16_t* __restrict__ WT) {
    __shared__ bf16_t tile[32][33];
    int t  = blockIdx.z;
    int k0 = blockIdx.x * 32;
    int n0 = blockIdx.y * 32;
    int lx = threadIdx.x & 31;
    int ly = threadIdx.x >> 5;
    const float* Wt = W + (size_t)t * D * D;
    #pragma unroll
    for (int i = 0; i < 32; i += 8)
        tile[ly + i][lx] = (bf16_t)Wt[(size_t)(k0 + ly + i) * D + (n0 + lx)];
    __syncthreads();
    bf16_t* WTt = WT + (size_t)t * D * D;
    #pragma unroll
    for (int i = 0; i < 32; i += 8)
        WTt[(size_t)(n0 + ly + i) * D + (k0 + lx)] = tile[lx][ly + i];
}

// ---------------------------------------------------------------------------
// Phase 1: out[b][t][:] = x[b][t][:] @ Wx[t]    (f32 result staged in d_out)
// grid = (S, B/32); block = 256 (8 waves). Wave w: M-tiles {0,1}, N-tiles {2w,2w+1}.
// ---------------------------------------------------------------------------
__global__ __launch_bounds__(256) void k_xwx(const float* __restrict__ x,
                                             const bf16_t* __restrict__ WxT,
                                             float* __restrict__ out) {
    __shared__ bf16_t xs[32 * LDA];
    const int t   = blockIdx.x;
    const int b0  = blockIdx.y * 32;
    const int tid = threadIdx.x;

    #pragma unroll 4
    for (int i = 0; i < 32; ++i) {
        float v = x[((size_t)(b0 + i) * S_LEN + t) * D + tid];
        xs[i * LDA + tid] = (bf16_t)v;
    }
    __syncthreads();

    const int wave = tid >> 5, lane = tid & 31;
    const int colBase = wave * 32;
    const bf16_t* Wt = WxT + (size_t)t * D * D;

    v8f acc[2][2] = {};
    #pragma unroll
    for (int kc = 0; kc < D; kc += 32) {
        v16bf bb0 = load_b(Wt, D, colBase,      kc, lane);
        v16bf bb1 = load_b(Wt, D, colBase + 16, kc, lane);
        v16bf a0  = load_a(xs, 0,  kc, lane);
        v16bf a1  = load_a(xs, 16, kc, lane);
        acc[0][0] = wmma_bf16(a0, bb0, acc[0][0]);
        acc[0][1] = wmma_bf16(a0, bb1, acc[0][1]);
        acc[1][0] = wmma_bf16(a1, bb0, acc[1][0]);
        acc[1][1] = wmma_bf16(a1, bb1, acc[1][1]);
    }

    const int colLo = lane & 15;
    const int rowHi = (lane >> 4) << 3;
    #pragma unroll
    for (int m = 0; m < 2; ++m)
        #pragma unroll
        for (int n = 0; n < 2; ++n)
            #pragma unroll
            for (int r = 0; r < 8; ++r) {
                int row = b0 + m * 16 + rowHi + r;
                int col = colBase + n * 16 + colLo;
                out[((size_t)row * S_LEN + t) * D + col] = acc[m][n][r];
            }
}

// ---------------------------------------------------------------------------
// Phase 2: recurrence. grid = B/16 WGs (persistent over all t).
// Wh[t] double-buffered in LDS via TDM (TENSORcnt), h ping-pong in LDS.
// out holds A[t] on entry, h_t on exit (read-then-overwrite by same thread).
// LDS: 2*256*264*2 (Wh) + 2*16*264*2 (h) = 287,232 B  (< 320KB/WGP, 1 WG/WGP)
// ---------------------------------------------------------------------------
__global__ __launch_bounds__(256) void k_recur(const bf16_t* __restrict__ WhT,
                                               float* __restrict__ out) {
    __shared__ bf16_t wh[2][256 * LDW];
    __shared__ bf16_t hs[2][16 * LDA];
    const int b0   = blockIdx.x * 16;
    const int tid  = threadIdx.x;
    const int wave = tid >> 5, lane = tid & 31;
    const int colBase = wave * 32;
    const int colLo = lane & 15;
    const int rowHi = (lane >> 4) << 3;

    // h_{-1} = 0; prime Wh[0] into buffer 0
    for (int i = tid; i < 16 * LDA; i += 256) hs[0][i] = (bf16_t)0.0f;
    stage_wh(wh[0], WhT, tid);
    stage_wait(tid);
    __syncthreads();

    for (int t = 0; t < S_LEN; ++t) {
        // Kick off the DMA for step t+1 while computing step t
        if (t + 1 < S_LEN)
            stage_wh(wh[(t + 1) & 1], WhT + (size_t)(t + 1) * D * D, tid);

        const bf16_t* Wl    = wh[t & 1];
        const bf16_t* hprev = hs[t & 1];
        bf16_t*       hnext = hs[(t + 1) & 1];

        v8f acc[2] = {};
        #pragma unroll
        for (int kc = 0; kc < D; kc += 32) {
            v16bf bb0 = load_b(Wl, LDW, colBase,      kc, lane);
            v16bf bb1 = load_b(Wl, LDW, colBase + 16, kc, lane);
            v16bf a   = load_a(hprev, 0, kc, lane);
            acc[0] = wmma_bf16(a, bb0, acc[0]);
            acc[1] = wmma_bf16(a, bb1, acc[1]);
        }

        // h = tanh(A_t + h_prev @ Wh[t]); f32 -> out, bf16 -> LDS
        #pragma unroll
        for (int n = 0; n < 2; ++n)
            #pragma unroll
            for (int r = 0; r < 8; ++r) {
                int row = rowHi + r;
                int col = colBase + n * 16 + colLo;
                size_t oi = ((size_t)(b0 + row) * S_LEN + t) * D + col;
                float v = tanhf(acc[n][r] + out[oi]);
                out[oi] = v;
                hnext[row * LDA + col] = (bf16_t)v;
            }

        stage_wait(tid);      // TDM for t+1 done (wave 0, TENSORcnt==0)
        __syncthreads();      // h + Wh buffers consistent for step t+1
    }
}

// ---------------------------------------------------------------------------
extern "C" void kernel_launch(void* const* d_in, const int* in_sizes, int n_in,
                              void* d_out, int out_size, void* d_ws, size_t ws_size,
                              hipStream_t stream) {
    const float* x  = (const float*)d_in[0];   // [B][S][D_IN]  f32
    const float* Wx = (const float*)d_in[1];   // [S][D_IN][D_OUT] f32
    const float* Wh = (const float*)d_in[2];   // [S][D_OUT][D_OUT] f32
    float* out = (float*)d_out;                // [B][S][D_OUT] f32

    bf16_t* WxT = (bf16_t*)d_ws;               // 64MB
    bf16_t* WhT = WxT + (size_t)S_LEN * D * D; // 64MB

    dim3 tb(256);

    dim3 tgT(D / 32, D / 32, S_LEN);
    k_transpose_cvt<<<tgT, tb, 0, stream>>>(Wx, WxT);
    k_transpose_cvt<<<tgT, tb, 0, stream>>>(Wh, WhT);

    k_xwx<<<dim3(S_LEN, 512 / 32), tb, 0, stream>>>(x, WxT, out);

    k_recur<<<dim3(512 / 16), tb, 0, stream>>>(WhT, out);
}
